// lstm_crf_69604239999374
// MI455X (gfx1250) — matrix operand order repaired and checked by reading, not compiled, not run on hardware
//
#include <hip/hip_runtime.h>
#include <math.h>
#include <stdint.h>

#define T_LEN   4096
#define EMBED   500
#define HID     500
#define G4H     2000      // 4*H gate width
#define NTAGS   16
#define START_TAG 14
#define STOP_TAG  15
#define NEGV    -10000.0f

// GEMM macro-tile: 32 rows (2 x 16) by 80 cols (5 x 16) per wave. 80*25 = 2000.
#define MT 2
#define NT 5

typedef __bf16 bf16_t;
typedef __attribute__((ext_vector_type(16))) __bf16 v16bf;
typedef __attribute__((ext_vector_type(8)))  float  v8f;

union FragBF { v16bf v; uint32_t u[8]; };

__device__ __forceinline__ float sigf(float x) { return 1.0f / (1.0f + __expf(-x)); }

// ---------------------------------------------------------------------------
// Embedding gather + f32->bf16 with K padding 500->512
// grid: (T_LEN), block: 256
__global__ void k_embed_gather(const int* __restrict__ sent,
                               const float* __restrict__ embed_w,
                               bf16_t* __restrict__ xout) {
    int t   = blockIdx.x;
    int tok = sent[t];
    const float* row = embed_w + (size_t)tok * EMBED;
    for (int k = threadIdx.x; k < 512; k += 256)
        xout[(size_t)t * 512 + k] = (k < EMBED) ? (bf16_t)row[k] : (bf16_t)0.0f;
}

// ---------------------------------------------------------------------------
// Generic f32 [rows,K] -> bf16 [rows,Kp] zero-padded converter
__global__ void k_conv_pad_bf16(const float* __restrict__ in,
                                bf16_t* __restrict__ out,
                                int rows, int K, int Kp) {
    int idx = blockIdx.x * blockDim.x + threadIdx.x;
    if (idx >= rows * Kp) return;
    int r = idx / Kp, k = idx - r * Kp;
    out[idx] = (k < K) ? (bf16_t)in[(size_t)r * K + k] : (bf16_t)0.0f;
}

// ---------------------------------------------------------------------------
// WMMA GEMM: out[t,n] = sum_k X[t,k]*W[n,k] + bih[n] + bhh[n]
// X: [T_LEN, KP] bf16 row-major, W: [G4H, KP] bf16 row-major, out: [T_LEN, G4H] f32
// grid: (T_LEN/32, 2000/80) = (128, 25), block: 32 (one wave per 32x80 C macro-tile)
// __launch_bounds__(32, 1): single wave per block, occupancy 1 -> full VGPR
// budget, no accumulator spills (10 x v8f acc + 7 fragments must stay resident).
// Fragment layouts as loaded here (wave32, contiguous k = 16h + s for A and B):
//   A 16x32 bf16: lane L<16 row M=L, VGPR v K={2v,2v+1};
//                 lanes>=16: K += 16 (same order as B).
//   B 32x16 bf16: lane L col N=L%16, VGPR v K={2v,2v+1}; lanes>=16: K += 16.
//   C/D f32: VGPR r -> row M=r (+8 for lanes>=16), col N=L%16.
template <int KP>
__global__ __launch_bounds__(32, 1)
void k_gemm_wmma_bf16(const bf16_t* __restrict__ X,
                      const bf16_t* __restrict__ W,
                      const float* __restrict__ bih,
                      const float* __restrict__ bhh,
                      float* __restrict__ out) {
    const int lane = threadIdx.x;
    const int half = (lane >> 4) & 1;     // 0: lanes 0-15, 1: lanes 16-31
    const int l16  = lane & 15;
    const int m0 = blockIdx.x * (16 * MT);
    const int n0 = blockIdx.y * (16 * NT);

    const uint32_t* Xr[MT];               // per M-subtile A row (dword = bf16 pair)
#pragma unroll
    for (int mi = 0; mi < MT; ++mi)
        Xr[mi] = (const uint32_t*)(X + (size_t)(m0 + 16 * mi + l16) * KP);
    const uint32_t* Wr[NT];               // per N-subtile B column
#pragma unroll
    for (int ni = 0; ni < NT; ++ni)
        Wr[ni] = (const uint32_t*)(W + (size_t)(n0 + 16 * ni + l16) * KP);

    v8f acc[MT][NT];
#pragma unroll
    for (int mi = 0; mi < MT; ++mi)
#pragma unroll
        for (int ni = 0; ni < NT; ++ni)
            acc[mi][ni] = (v8f){};

#pragma unroll 1
    for (int k0 = 0; k0 < KP; k0 += 32) {
        FragBF a[MT], b[NT];
#pragma unroll
        for (int mi = 0; mi < MT; ++mi)
#pragma unroll
            for (int v = 0; v < 8; ++v) {
                int kA = k0 + 2 * v + half * 16;
                a[mi].u[v] = Xr[mi][kA >> 1];
            }
#pragma unroll
        for (int ni = 0; ni < NT; ++ni)
#pragma unroll
            for (int v = 0; v < 8; ++v) {
                int kB = k0 + 2 * v + half * 16;
                b[ni].u[v] = Wr[ni][kB >> 1];
            }
#pragma unroll
        for (int mi = 0; mi < MT; ++mi)
#pragma unroll
            for (int ni = 0; ni < NT; ++ni)
                acc[mi][ni] = __builtin_amdgcn_wmma_f32_16x16x32_bf16(
                        false, a[mi].v, false, b[ni].v,
                        (short)0, acc[mi][ni], false, false);
    }

#pragma unroll
    for (int ni = 0; ni < NT; ++ni) {
        const int n = n0 + 16 * ni + l16;         // fixed column per lane/subtile
        const float bias = bih[n] + bhh[n];
#pragma unroll
        for (int mi = 0; mi < MT; ++mi)
#pragma unroll
            for (int r = 0; r < 8; ++r) {
                int row = m0 + 16 * mi + r + half * 8;
                out[(size_t)row * G4H + n] = acc[mi][ni][r] + bias;
            }
    }
}

// ---------------------------------------------------------------------------
// Sequential LSTM recurrence, BOTH directions of one layer concurrently:
// blockIdx.x = direction (0=fwd, 1=bwd), each block persists on its own WGP.
// XWf/XWb: [T_LEN, G4H] f32 gate pre-activations (x@wih.T + bias), per direction.
// whh: [2, G4H, HID] f32 (L2-resident). h0/c0: [2, HID] for this layer.
// hcat: [T_LEN, 2*HID] f32; direction d writes columns [d*HID, (d+1)*HID).
// block: 1024 threads (32 waves).
__global__ void k_lstm_rec_pair(const float* __restrict__ XWf,
                                const float* __restrict__ XWb,
                                const float* __restrict__ whh,
                                const float* __restrict__ h0,
                                const float* __restrict__ c0,
                                float* __restrict__ hcat) {
    __shared__ __align__(16) float sh_h[512];
    __shared__ float sh_c[512];
    __shared__ float sh_g[2048];
    const int tid = threadIdx.x;
    const int dir = blockIdx.x;
    const float* XW    = dir ? XWb : XWf;
    const float* whh_d = whh + (size_t)dir * G4H * HID;
    const int col_off  = dir * HID;

    if (tid < HID) { sh_h[tid] = h0[dir * HID + tid]; sh_c[tid] = c0[dir * HID + tid]; }
    __syncthreads();

    for (int s = 0; s < T_LEN; ++s) {
        const int t = dir ? (T_LEN - 1 - s) : s;
        const float* xw = XW + (size_t)t * G4H;

        // Prefetch next timestep's gate row into cache (global_prefetch_b8)
        int tn = dir ? (t - 1) : (t + 1);
        if (tn >= 0 && tn < T_LEN)
            __builtin_prefetch(XW + (size_t)tn * G4H + (tid & 63) * 32, 0, 1);

        // Gate pre-activations: g[j] = xw[j] + dot(whh[j,:], h)
        for (int j = tid; j < G4H; j += 1024) {
            const float4* w4 = (const float4*)(whh_d + (size_t)j * HID);
            const float4* h4 = (const float4*)sh_h;
            float acc = xw[j];
#pragma unroll 5
            for (int k = 0; k < HID / 4; ++k) {
                float4 w = w4[k], h = h4[k];
                acc += w.x * h.x + w.y * h.y + w.z * h.z + w.w * h.w;
            }
            sh_g[j] = acc;
        }
        __syncthreads();

        // Cell update (torch gate order i,f,g,o)
        if (tid < HID) {
            float gi = sh_g[tid];
            float gf = sh_g[HID + tid];
            float gg = sh_g[2 * HID + tid];
            float go = sh_g[3 * HID + tid];
            float cn = sigf(gf) * sh_c[tid] + sigf(gi) * tanhf(gg);
            float hn = sigf(go) * tanhf(cn);
            sh_c[tid] = cn;
            sh_h[tid] = hn;
            hcat[(size_t)t * (2 * HID) + col_off + tid] = hn;
        }
        __syncthreads();
    }
}

// ---------------------------------------------------------------------------
// feats[t,tag] = dot(hcat[t,:], w_out[tag,:]) + b_out[tag]
// grid: (T_LEN), block: 32 (16 active lanes)
__global__ void k_feats(const float* __restrict__ hcat,
                        const float* __restrict__ w_out,
                        const float* __restrict__ b_out,
                        float* __restrict__ feats) {
    int t = blockIdx.x, tag = threadIdx.x;
    if (tag >= NTAGS) return;
    const float* h = hcat + (size_t)t * (2 * HID);
    const float* w = w_out + (size_t)tag * (2 * HID);
    float acc = b_out[tag];
    for (int k = 0; k < 2 * HID; ++k) acc += h[k] * w[k];
    feats[t * NTAGS + tag] = acc;
}

// ---------------------------------------------------------------------------
// Viterbi decode (single wave). out[0]=best_score, out[1..T]=path (as float).
// Faithful to reference backtrace: path = [bptr[1][bt], ..., bptr[T-1][bt], bt].
__global__ void k_viterbi(const float* __restrict__ feats,
                          const float* __restrict__ trans,
                          int* __restrict__ bptr,
                          float* __restrict__ out) {
    __shared__ float sc[NTAGS];
    const int tid = threadIdx.x;
    if (tid < NTAGS) sc[tid] = (tid == START_TAG) ? 0.0f : NEGV;
    __syncthreads();

    for (int t = 0; t < T_LEN; ++t) {
        float best = -3.4e38f; int bj = 0;
        if (tid < NTAGS) {
            for (int p = 0; p < NTAGS; ++p) {         // first-max tie-break
                float v = sc[p] + trans[tid * NTAGS + p];
                if (v > best) { best = v; bj = p; }
            }
        }
        __syncthreads();
        if (tid < NTAGS) {
            sc[tid] = best + feats[t * NTAGS + tid];
            bptr[t * NTAGS + tid] = bj;
        }
        __syncthreads();
    }

    if (tid == 0) {
        float bs = -3.4e38f; int bt = 0;
        for (int p = 0; p < NTAGS; ++p) {
            float v = sc[p] + trans[STOP_TAG * NTAGS + p];
            if (v > bs) { bs = v; bt = p; }
        }
        out[0] = bs;
        for (int i = 0; i < T_LEN - 1; ++i)
            out[1 + i] = (float)bptr[(i + 1) * NTAGS + bt];
        out[T_LEN] = (float)bt;
    }
}

// ---------------------------------------------------------------------------
extern "C" void kernel_launch(void* const* d_in, const int* in_sizes, int n_in,
                              void* d_out, int out_size, void* d_ws, size_t ws_size,
                              hipStream_t stream) {
    const int*   sent    = (const int*)  d_in[0];
    const float* embed_w = (const float*)d_in[1];
    const float* wih0    = (const float*)d_in[2];
    const float* whh0    = (const float*)d_in[3];
    const float* bih0    = (const float*)d_in[4];
    const float* bhh0    = (const float*)d_in[5];
    const float* wih1    = (const float*)d_in[6];
    const float* whh1    = (const float*)d_in[7];
    const float* bih1    = (const float*)d_in[8];
    const float* bhh1    = (const float*)d_in[9];
    const float* w_out   = (const float*)d_in[10];
    const float* b_out   = (const float*)d_in[11];
    const float* trans   = (const float*)d_in[12];
    const float* h0      = (const float*)d_in[13];
    const float* c0      = (const float*)d_in[14];
    float* out = (float*)d_out;

    // Workspace carve-up (~108 MB total)
    char* ws = (char*)d_ws;
    size_t off = 0;
    auto carve = [&](size_t bytes) -> char* {
        char* p = ws + off;
        off = (off + bytes + 255) & ~(size_t)255;
        return p;
    };
    bf16_t* xin  = (bf16_t*)carve((size_t)T_LEN * 512 * 2);        // layer-0 input, bf16 padded
    bf16_t* wbf0 = (bf16_t*)carve((size_t)2 * G4H * 512 * 2);      // wih0 bf16 padded
    bf16_t* wbf1 = (bf16_t*)carve((size_t)2 * G4H * 1024 * 2);     // wih1 bf16 padded
    float*  XWa  = (float*) carve((size_t)T_LEN * G4H * 4);        // fwd gate pre-acts
    float*  XWb  = (float*) carve((size_t)T_LEN * G4H * 4);        // bwd gate pre-acts
    float*  hcat = (float*) carve((size_t)T_LEN * 2 * HID * 4);    // bi-dir hidden concat
    bf16_t* hbf  = (bf16_t*)carve((size_t)T_LEN * 1024 * 2);       // layer-1 input bf16 padded
    float*  feats= (float*) carve((size_t)T_LEN * NTAGS * 4);
    int*    bptr = (int*)   carve((size_t)T_LEN * NTAGS * 4);

    const dim3 gemm_grid(T_LEN / (16 * MT), G4H / (16 * NT));  // 128 x 25 macro-tiles

    // Embedding + weight conversions (run every call; deterministic)
    k_embed_gather<<<T_LEN, 256, 0, stream>>>(sent, embed_w, xin);
    {
        int n = 2 * G4H * 512;
        k_conv_pad_bf16<<<(n + 255) / 256, 256, 0, stream>>>(wih0, wbf0, 2 * G4H, 500, 512);
    }
    {
        int n = 2 * G4H * 1024;
        k_conv_pad_bf16<<<(n + 255) / 256, 256, 0, stream>>>(wih1, wbf1, 2 * G4H, 1000, 1024);
    }

    // ---- Layer 0: both direction GEMMs, then both recurrences concurrently ----
    k_gemm_wmma_bf16<512><<<gemm_grid, 32, 0, stream>>>(xin, wbf0, bih0, bhh0, XWa);
    k_gemm_wmma_bf16<512><<<gemm_grid, 32, 0, stream>>>(xin, wbf0 + (size_t)G4H * 512,
                                                        bih0 + G4H, bhh0 + G4H, XWb);
    k_lstm_rec_pair<<<2, 1024, 0, stream>>>(XWa, XWb, whh0, h0, c0, hcat);

    // ---- Layer 1 ----
    {
        int n = T_LEN * 1024;
        k_conv_pad_bf16<<<(n + 255) / 256, 256, 0, stream>>>(hcat, hbf, T_LEN, 1000, 1024);
    }
    k_gemm_wmma_bf16<1024><<<gemm_grid, 32, 0, stream>>>(hbf, wbf1, bih1, bhh1, XWa);
    k_gemm_wmma_bf16<1024><<<gemm_grid, 32, 0, stream>>>(hbf, wbf1 + (size_t)G4H * 1024,
                                                         bih1 + G4H, bhh1 + G4H, XWb);
    k_lstm_rec_pair<<<2, 1024, 0, stream>>>(XWa, XWb, whh1, h0 + 2 * HID, c0 + 2 * HID, hcat);

    // ---- CRF head ----
    k_feats<<<T_LEN, 32, 0, stream>>>(hcat, w_out, b_out, feats);
    k_viterbi<<<1, 32, 0, stream>>>(feats, trans, bptr, out);
}
